// SimAttention_21595095564588
// MI455X (gfx1250) — matrix operand order repaired and checked by reading, not compiled, not run on hardware
//
#include <hip/hip_runtime.h>
#include <hip/hip_bf16.h>
#include <math.h>

// ---- problem constants (B=2,H=16,S=2048,D=64) ----
#define SEQ   2048
#define DIM   64
#define NHEAD 32            // B*H
#define NROWS (NHEAD * SEQ) // rows per tensor
#define NEGV  (-10000.0f)

typedef __attribute__((ext_vector_type(16))) _Float16 v16h;
typedef __attribute__((ext_vector_type(8)))  _Float16 v8h;
typedef __attribute__((ext_vector_type(8)))  float    v8f;

// ---------------------------------------------------------------------------
// Kernel 1: prep. One wave per row-job:
//   job < NROWS        : q row  -> rq = 1/(8*sum|q|)   (1/sqrt(64) folded in)
//   job < 2*NROWS      : k row  -> kn16[row][d] = f16(k/sum|k|)   (row major)
//   else               : v row  -> vT16[head][d][k] = f16(v)      (transposed)
// ---------------------------------------------------------------------------
__global__ __launch_bounds__(256) void prep_kernel(
    const float* __restrict__ q, const float* __restrict__ k,
    const float* __restrict__ v, float* __restrict__ rq,
    _Float16* __restrict__ kn16, _Float16* __restrict__ vT16)
{
    const int job  = blockIdx.x * 8 + (threadIdx.x >> 5);
    const int lane = threadIdx.x & 31;

    if (job < NROWS) {
        const float* pr = q + (size_t)job * DIM + lane * 2;
        float s = fabsf(pr[0]) + fabsf(pr[1]);
        #pragma unroll
        for (int off = 16; off; off >>= 1) s += __shfl_xor(s, off, 32);
        if (lane == 0) rq[job] = 0.125f / s;
    } else if (job < 2 * NROWS) {
        const int row = job - NROWS;
        const float* pr = k + (size_t)row * DIM + lane * 2;
        const float a = pr[0], b = pr[1];
        float s = fabsf(a) + fabsf(b);
        #pragma unroll
        for (int off = 16; off; off >>= 1) s += __shfl_xor(s, off, 32);
        const float inv = 1.0f / s;   // butterfly leaves full sum in all lanes
        kn16[(size_t)row * DIM + lane * 2]     = (_Float16)(a * inv);
        kn16[(size_t)row * DIM + lane * 2 + 1] = (_Float16)(b * inv);
    } else {
        const int row = job - 2 * NROWS;
        const int h = row / SEQ, kk = row % SEQ;
        const float* pr = v + (size_t)row * DIM + lane * 2;
        _Float16* dst = vT16 + (size_t)h * DIM * SEQ + kk;
        dst[(size_t)(lane * 2)     * SEQ] = (_Float16)pr[0];
        dst[(size_t)(lane * 2 + 1) * SEQ] = (_Float16)pr[1];
    }
}

// ---------------------------------------------------------------------------
// Kernel 2: per-head suffix sums of V at 16-row granularity (f32 exact).
// st16[head][t][d] = sum_{kk >= 16t} v[head][kk][d], t = 0..128 (st16[128]=0)
// ---------------------------------------------------------------------------
__global__ __launch_bounds__(64) void suffix_kernel(
    const float* __restrict__ v, float* __restrict__ st16)
{
    const int head = blockIdx.x;
    const int d    = threadIdx.x; // 0..63
    const float* vb = v + (size_t)head * SEQ * DIM;
    float* st = st16 + (size_t)head * 129 * DIM;
    float run = 0.0f;
    st[128 * DIM + d] = 0.0f;
    for (int t = 127; t >= 0; --t) {
        #pragma unroll 4
        for (int j = 15; j >= 0; --j) run += vb[(size_t)(16 * t + j) * DIM + d];
        st[t * DIM + d] = run;
    }
}

// ---------------------------------------------------------------------------
// Kernel 3: main attention. grid = (SEQ/128, NHEAD), block = 256 (8 waves).
// 64-wide K tiles, double-buffered in LDS via GLOBAL_LOAD_ASYNC_TO_LDS_B128
// (ASYNCcnt), 16 WMMAs per barrier pair; causal-future tiles folded in via
// NEG * suffix sums. Wave w owns q-rows [blk*128 + 16w, +16).
// ---------------------------------------------------------------------------
__global__ __launch_bounds__(256) void simattn_main(
    const float* __restrict__ q, const float* __restrict__ rq,
    const _Float16* __restrict__ kn16, const _Float16* __restrict__ vT16,
    const float* __restrict__ st16, float* __restrict__ out)
{
    __shared__ __attribute__((aligned(32))) _Float16 knA[2][64 * 64]; // [k][d]
    __shared__ __attribute__((aligned(32))) _Float16 vA [2][64 * 64]; // [d][k]
    __shared__ __attribute__((aligned(32))) _Float16 pst[8][16 * 64]; // per-wave P

    const int head = blockIdx.y;
    const int tid  = threadIdx.x;
    const int wave = tid >> 5;
    const int lane = tid & 31;
    const int g    = lane >> 4;   // half-group (0/1)
    const int n    = lane & 15;   // N / M index within group

    const size_t base = (size_t)head * SEQ * DIM;
    const int q0blk  = blockIdx.x * 128;
    const int q0w    = q0blk + wave * 16;
    const int qt     = q0w >> 4;           // 16-row q tile index
    const int kt_d   = qt >> 2;            // my diagonal 64-wide k tile
    const int kt_max = (q0blk + 127) >> 6; // block-uniform loop bound

    const _Float16* knh = kn16 + base;     // [k][d], row stride DIM
    const _Float16* vth = vT16 + base;     // [d][k], row stride SEQ

    // Issue async staging of 64-wide k tile `kt` into buffer `bf`:
    // 8KB K tile + 8KB V tile, two b128 async loads each per thread.
    auto issue = [&](int kt, int bf) {
        const int k0 = kt * 64;
        #pragma unroll
        for (int i = 0; i < 2; ++i) {               // K tile: contiguous 8KB
            const int ch = tid + i * 256;           // 16B chunk id, 0..511
            uint64_t ga = (uint64_t)(uintptr_t)(knh + (size_t)k0 * DIM) + (uint32_t)(ch * 16);
            uint32_t la = (uint32_t)(uintptr_t)(&knA[bf][0]) + (uint32_t)(ch * 16);
            asm volatile("global_load_async_to_lds_b128 %0, %1, off"
                         :: "v"(la), "v"(ga) : "memory");
        }
        #pragma unroll
        for (int i = 0; i < 2; ++i) {               // V tile: 64 rows x 128B
            const int ch = tid + i * 256;           // 0..511
            const int r = ch >> 3, c = ch & 7;
            uint64_t ga = (uint64_t)(uintptr_t)(vth + (size_t)r * SEQ + k0) + (uint32_t)(c * 16);
            uint32_t la = (uint32_t)(uintptr_t)(&vA[bf][r * 64]) + (uint32_t)(c * 16);
            asm volatile("global_load_async_to_lds_b128 %0, %1, off"
                         :: "v"(la), "v"(ga) : "memory");
        }
    };

    // ---- my Q tile in WMMA A-layout (16x32 f16, 2 K-steps over D=64) ------
    const int   mrow = q0w + n;            // A-matrix row = lane%16
    const float sq   = rq[head * SEQ + mrow];
    v16h aq[2];
    #pragma unroll
    for (int s = 0; s < 2; ++s) {
        const float* qr = q + base + (size_t)mrow * DIM + s * 32;
        #pragma unroll
        for (int h = 0; h < 8; ++h) {
            aq[s][h]     = (_Float16)(qr[16 * g + h] * sq);     // K = 16g+h
            aq[s][8 + h] = (_Float16)(qr[16 * g + 8 + h] * sq); // K = 16g+8+h
        }
    }

    v8f acc[4] = {}; // 16 q-rows x 64 d-cols, f32

    issue(0, 0);
    for (int kt = 0; kt <= kt_max; ++kt) {
        const int bf = kt & 1;
        asm volatile("s_wait_asynccnt 0x0" ::: "memory"); // my tile-kt copies landed
        __syncthreads();                                  // everyone's landed
        if (kt < kt_max) issue(kt + 1, bf ^ 1);           // prefetch next tile

        const int act = (kt <= kt_d);
        if (__builtin_amdgcn_readfirstlane(act)) {
            const int k0 = kt * 64;
            // ---- scores: S[16q x 64k] as four 16x16 N-halves, chained K=32 x2
            #pragma unroll
            for (int nh = 0; nh < 4; ++nh) {
                v8f c = {};
                const int krow = nh * 16 + n;   // B column = k index
                #pragma unroll
                for (int s = 0; s < 2; ++s) {
                    const v16h bk = *(const v16h*)(&knA[bf][krow * 64 + s * 32 + 16 * g]);
                    c = __builtin_amdgcn_wmma_f32_16x16x32_f16(
                            false, aq[s], false, bk, (short)0, c, false, false);
                }
                if (__builtin_amdgcn_readfirstlane((int)(kt == kt_d))) {
                    const int kcol = k0 + nh * 16 + n; // causal mask on diag tile
                    #pragma unroll
                    for (int r = 0; r < 8; ++r) {
                        const int m = r + 8 * g;
                        if (kcol > q0w + m) c[r] = NEGV;
                    }
                }
                #pragma unroll
                for (int r = 0; r < 8; ++r) {   // P quarter -> per-wave LDS (f16)
                    const int m = r + 8 * g;
                    pst[wave][m * 64 + nh * 16 + n] = (_Float16)c[r];
                }
            }
        }
        __syncthreads(); // make P stores visible for A-layout reload

        if (__builtin_amdgcn_readfirstlane(act)) {
            v16h ap[2]; // rebuild P in A-layout (16x64, 2 K-steps)
            #pragma unroll
            for (int s = 0; s < 2; ++s) {
                const v8h lo = *(const v8h*)(&pst[wave][n * 64 + s * 32 + 16 * g]);
                const v8h hi = *(const v8h*)(&pst[wave][n * 64 + s * 32 + 16 * g + 8]);
                #pragma unroll
                for (int h = 0; h < 8; ++h) { ap[s][h] = lo[h]; ap[s][8 + h] = hi[h]; }
            }
            #pragma unroll
            for (int nb = 0; nb < 4; ++nb) { // P(16x64) x V(64x64), chained K=32 x2
                const int d = nb * 16 + n;
                #pragma unroll
                for (int s = 0; s < 2; ++s) {
                    const v16h bv = *(const v16h*)(&vA[bf][d * 64 + s * 32 + 16 * g]);
                    acc[nb] = __builtin_amdgcn_wmma_f32_16x16x32_f16(
                                  false, ap[s], false, bv, (short)0, acc[nb], false, false);
                }
            }
        }
    }

    // ---- epilogue: add NEG * suffix of fully-future V rows, write out -----
    const int fidx = (kt_d + 1) * 4;                 // first fully-future 16-tile
    const float* sthead = st16 + (size_t)head * 129 * DIM;
    #pragma unroll
    for (int nb = 0; nb < 4; ++nb) {
        const int d = nb * 16 + n;
        const float suf = sthead[fidx * DIM + d] * NEGV;
        #pragma unroll
        for (int r = 0; r < 8; ++r) {
            const int m = r + 8 * g;
            out[base + (size_t)(q0w + m) * DIM + d] = acc[nb][r] + suf;
        }
    }
}

// ---------------------------------------------------------------------------
extern "C" void kernel_launch(void* const* d_in, const int* in_sizes, int n_in,
                              void* d_out, int out_size, void* d_ws, size_t ws_size,
                              hipStream_t stream) {
    const float* q = (const float*)d_in[0];
    const float* k = (const float*)d_in[1];
    const float* v = (const float*)d_in[2];
    // d_in[3] is the causal mask; causality is baked into the kernels.
    float* out = (float*)d_out;

    // workspace layout (~18.1 MB):
    //   rq   : NROWS f32
    //   st16 : NHEAD*129*64 f32
    //   kn16 : NROWS*64 f16   (scaled K, row major)
    //   vT16 : NROWS*64 f16   (V transposed per head, [d][k])
    float* rq   = (float*)d_ws;
    float* st16 = rq + NROWS;
    _Float16* kn16 = (_Float16*)(st16 + (size_t)NHEAD * 129 * DIM);
    _Float16* vT16 = kn16 + (size_t)NROWS * DIM;

    prep_kernel<<<(3 * NROWS) / 8, 256, 0, stream>>>(q, k, v, rq, kn16, vT16);
    suffix_kernel<<<NHEAD, 64, 0, stream>>>(v, st16);

    dim3 grid(SEQ / 128, NHEAD);
    simattn_main<<<grid, 256, 0, stream>>>(q, rq, kn16, vT16, st16, out);
}
